// Allegro_7275674599897
// MI455X (gfx1250) — matrix operand order repaired and checked by reading, not compiled, not run on hardware
//
#include <hip/hip_runtime.h>
#include <hip/hip_bf16.h>

typedef _Float16 h16;
typedef __attribute__((ext_vector_type(16))) _Float16 v16h;
typedef __attribute__((ext_vector_type(8)))  _Float16 v8h;
typedef __attribute__((ext_vector_type(8)))  float    v8f;

#define WAVES_PER_BLOCK 2
#define M_TILES 2              // 32 edges per wave = 2 WMMA M-tiles
#define EDGES_PER_WAVE 32

// Packed-weight offsets in halves inside d_ws (all multiples of 512 halves = 1KB)
static constexpr int OFF_WINIT = 0;                       // 160x128 = 20480
static constexpr int OFF_W1    = 20480;                   // 3 x 288x64 = 3*18432
static constexpr int OFF_W2    = 75776;                   // 3 x 64x64  = 3*4096
static constexpr int OFF_W3    = 88064;                   // 3 x 64x128 = 3*8192
static constexpr int OFF_TP1   = 112640;                  // 3 x 32x64  = 3*2048
static constexpr int OFF_TP2   = 118784;                  // 3 x 64x128 = 3*8192
static constexpr int OFF_HW1   = 143360;                  // 128x64 = 8192
static constexpr int OFF_HW2   = 151552;                  // 64x32  = 2048
// total = 153600 halves = 300 KiB

static __device__ __forceinline__ v8f wmma_f16(v16h a, v16h b, v8f c) {
  return __builtin_amdgcn_wmma_f32_16x16x32_f16(false, a, false, b, (short)0, c,
                                                false, false);
}

// Within-wave LDS producer->consumer sync (waves are independent; no block barrier).
static __device__ __forceinline__ void wave_lds_sync() {
  __builtin_amdgcn_wave_barrier();
  asm volatile("s_wait_dscnt 0" ::: "memory");
  __builtin_amdgcn_wave_barrier();
}

// A-fragment (16x32 f16) from LDS tile, row = edge (M), K starts at column kc.
// ISA layout: lane<16 -> M=lane, halves 0..7 = K+0..7, halves 8..15 = K+16..23;
//             lane>=16 -> M=lane-16, halves 0..7 = K+8..15, halves 8..15 = K+24..31.
static __device__ __forceinline__ v16h load_a_frag(const h16* base, int stride,
                                                   int kc, int lane) {
  const int m = lane & 15;
  const h16* p = base + m * stride + kc + ((lane & 16) ? 8 : 0);
  union { v16h v; v8h h[2]; } u;
  u.h[0] = *(const v8h*)(p);
  u.h[1] = *(const v8h*)(p + 16);
  return u.v;
}

// B-fragment (32x16 f16) from packed global weights: 16 contiguous halves/lane.
static __device__ __forceinline__ v16h load_b_frag(const h16* base, int tile_idx,
                                                   int lane) {
  return *(const v16h*)(base + (size_t)tile_idx * 512 + lane * 16);
}

static __device__ __forceinline__ float silu_f(float x) {
  return x / (1.0f + __expf(-x));
}

// ---------------- prepass: pack f32 weights into WMMA B-fragment order ----------
__global__ void pack_wmma_b(const float* __restrict__ src, h16* __restrict__ dst,
                            int K_ref, int N, int K_pad, int row_off) {
  const int total = K_pad * N;
  for (int p = blockIdx.x * blockDim.x + threadIdx.x; p < total;
       p += gridDim.x * blockDim.x) {
    const int tile = p >> 9;
    const int rem  = p & 511;
    const int lane = rem >> 4;
    const int i    = rem & 15;
    const int ntn  = N >> 4;
    const int kt   = tile / ntn;
    const int nt   = tile - kt * ntn;
    const int n    = nt * 16 + (lane & 15);
    const int k    = kt * 32 + ((lane & 16) ? 8 : 0) + (i & 7) + ((i & 8) ? 16 : 0);
    const int ks   = k - row_off;
    float v = (ks >= 0 && ks < K_ref) ? src[ks * N + n] : 0.0f;
    dst[p] = (h16)v;
  }
}

__global__ void zero_out_kernel(float* out) { out[0] = 0.0f; }

__global__ void node_energy_kernel(const int* __restrict__ an,
                                   const float* __restrict__ ae,
                                   float* __restrict__ out, int N) {
  float s = 0.0f;
  for (int i = blockIdx.x * blockDim.x + threadIdx.x; i < N;
       i += gridDim.x * blockDim.x)
    s += ae[an[i]];
  #pragma unroll
  for (int off = 16; off > 0; off >>= 1) s += __shfl_down(s, off);
  if ((threadIdx.x & 31) == 0) atomicAdd(out, s);
}

// ---------------- fused per-edge model: 1 wave = 32 edges (2 M-tiles) ------------
__global__ __launch_bounds__(64) void allegro_fused(
    const int* __restrict__ atomic_numbers, const float* __restrict__ pos,
    const int* __restrict__ edge_index, const float* __restrict__ node_emb,
    const h16* __restrict__ ws, const float* __restrict__ b_init,
    const float* __restrict__ b1, const float* __restrict__ b2,
    const float* __restrict__ b3, const float* __restrict__ ln_g,
    const float* __restrict__ ln_b, const float* __restrict__ tpb1,
    const float* __restrict__ tpb2, const float* __restrict__ hb1,
    const float* __restrict__ hb2, const float* __restrict__ hw3,
    const float* __restrict__ hb3, float* __restrict__ out, int N, int E) {
  // LDS tile per wave: 32 edges x 288 halves:
  // [0..127]=ef  [128..191]=ni  [192..255]=nj  [256..263]=rbf  [264..272]=sh  [273..287]=0
  __shared__ __align__(32) h16 s_tile[WAVES_PER_BLOCK][EDGES_PER_WAVE * 288];
  __shared__ __align__(32) h16 s_h[WAVES_PER_BLOCK][EDGES_PER_WAVE * 64];

  const int lane = threadIdx.x & 31;
  const int wave = threadIdx.x >> 5;
  const int n15  = lane & 15;
  const int mhi  = (lane & 16) ? 8 : 0;  // C-frag row offset for upper half-wave
  h16* tile = s_tile[wave];
  h16* hst  = s_h[wave];

  const long long tileIdx = (long long)blockIdx.x * WAVES_PER_BLOCK + wave;
  const long long base_e  = tileIdx * EDGES_PER_WAVE;

  // ---------------- stage per-edge inputs (one edge per lane) ----------------
  {
    long long e = base_e + lane;
    if (e >= E) e = (long long)E - 1;
    const int r0 = edge_index[e];
    const int c0 = edge_index[(long long)E + e];
    const int zi = atomic_numbers[r0];
    const int zj = atomic_numbers[c0];
    const float* nip = node_emb + zi * 64;
    const float* njp = node_emb + zj * 64;
    h16* trow = tile + lane * 288;
    #pragma unroll
    for (int j8 = 0; j8 < 8; ++j8) {  // 8 halves per step, packed b128 stores
      const float4 a0 = *(const float4*)(nip + j8 * 8);
      const float4 a1 = *(const float4*)(nip + j8 * 8 + 4);
      const float4 b0 = *(const float4*)(njp + j8 * 8);
      const float4 b1v = *(const float4*)(njp + j8 * 8 + 4);
      union { v8h v; h16 h[8]; } pa, pb;
      pa.h[0] = (h16)a0.x; pa.h[1] = (h16)a0.y; pa.h[2] = (h16)a0.z;
      pa.h[3] = (h16)a0.w; pa.h[4] = (h16)a1.x; pa.h[5] = (h16)a1.y;
      pa.h[6] = (h16)a1.z; pa.h[7] = (h16)a1.w;
      pb.h[0] = (h16)b0.x; pb.h[1] = (h16)b0.y; pb.h[2] = (h16)b0.z;
      pb.h[3] = (h16)b0.w; pb.h[4] = (h16)b1v.x; pb.h[5] = (h16)b1v.y;
      pb.h[6] = (h16)b1v.z; pb.h[7] = (h16)b1v.w;
      *(v8h*)(trow + 128 + j8 * 8) = pa.v;
      *(v8h*)(trow + 192 + j8 * 8) = pb.v;
    }
    // geometry: rbf + spherical harmonics
    const float dx = pos[c0 * 3 + 0] - pos[r0 * 3 + 0];
    const float dy = pos[c0 * 3 + 1] - pos[r0 * 3 + 1];
    const float dz = pos[c0 * 3 + 2] - pos[r0 * 3 + 2];
    const float d  = sqrtf(dx * dx + dy * dy + dz * dz);
    const float env =
        (d < 5.0f) ? 0.5f * (__cosf(d * 0.6283185307f) + 1.0f) : 0.0f;
    const float centers[8] = {0.98078528f,  0.83146961f,  0.55557023f,
                              0.19509032f,  -0.19509032f, -0.55557023f,
                              -0.83146961f, -0.98078528f};
    #pragma unroll
    for (int k = 0; k < 8; ++k) {
      const float diff = (d - centers[k] * 5.0f) * 1.6f;  // / width (0.625)
      trow[256 + k] = (h16)(__expf(-0.5f * diff * diff) * env);
    }
    const float rr  = fmaxf(d, 1e-8f);
    const float inv = 1.0f / rr;
    const float x = dx * inv, y = dy * inv, z = dz * inv;
    const float c1 = 0.4886025119f, c2 = 1.0925484306f;
    trow[264] = (h16)0.2820947918f;
    trow[265] = (h16)(-c1 * y);
    trow[266] = (h16)(c1 * z);
    trow[267] = (h16)(-c1 * x);
    trow[268] = (h16)(c2 * x * y);
    trow[269] = (h16)(-c2 * y * z);
    trow[270] = (h16)(0.9461746958f * z * z - 0.3153915653f * (x * x + y * y));
    trow[271] = (h16)(-c2 * x * z);
    trow[272] = (h16)(0.5462742153f * (x * x - y * y));
    #pragma unroll
    for (int c = 273; c < 288; ++c) trow[c] = (h16)0.0f;  // K padding must be 0
  }
  wave_lds_sync();

  // ---------------- ef = [ni|nj|rbf|sh] @ Winit + b_init ----------------
  {
    v8f acc[M_TILES][8] = {{{}, {}, {}, {}, {}, {}, {}, {}},
                           {{}, {}, {}, {}, {}, {}, {}, {}}};
    #pragma unroll
    for (int kt = 0; kt < 5; ++kt) {
      const v16h A0 = load_a_frag(tile, 288, 128 + kt * 32, lane);
      const v16h A1 = load_a_frag(tile + 16 * 288, 288, 128 + kt * 32, lane);
      #pragma unroll
      for (int nt = 0; nt < 8; ++nt) {
        const v16h B = load_b_frag(ws + OFF_WINIT, kt * 8 + nt, lane);
        acc[0][nt] = wmma_f16(A0, B, acc[0][nt]);
        acc[1][nt] = wmma_f16(A1, B, acc[1][nt]);
      }
    }
    #pragma unroll
    for (int mt = 0; mt < M_TILES; ++mt) {
      #pragma unroll
      for (int nt = 0; nt < 8; ++nt) {
        const float bb = b_init[nt * 16 + n15];
        #pragma unroll
        for (int r = 0; r < 8; ++r)
          tile[(mt * 16 + r + mhi) * 288 + nt * 16 + n15] =
              (h16)(acc[mt][nt][r] + bb);
      }
    }
  }
  wave_lds_sync();

  // ---------------- 3 interaction layers ----------------
  for (int l = 0; l < 3; ++l) {
    const h16* W1 = ws + OFF_W1 + l * 18432;
    const h16* W2 = ws + OFF_W2 + l * 4096;
    const h16* W3 = ws + OFF_W3 + l * 8192;
    const h16* T1 = ws + OFF_TP1 + l * 2048;
    const h16* T2 = ws + OFF_TP2 + l * 8192;

    // h = silu([ef|ni|nj|rbf] @ w1 + b1)   (K=288 incl. zero-weight padding)
    {
      v8f a4[M_TILES][4] = {{{}, {}, {}, {}}, {{}, {}, {}, {}}};
      #pragma unroll
      for (int kt = 0; kt < 9; ++kt) {
        const v16h A0 = load_a_frag(tile, 288, kt * 32, lane);
        const v16h A1 = load_a_frag(tile + 16 * 288, 288, kt * 32, lane);
        #pragma unroll
        for (int nt = 0; nt < 4; ++nt) {
          const v16h B = load_b_frag(W1, kt * 4 + nt, lane);
          a4[0][nt] = wmma_f16(A0, B, a4[0][nt]);
          a4[1][nt] = wmma_f16(A1, B, a4[1][nt]);
        }
      }
      #pragma unroll
      for (int mt = 0; mt < M_TILES; ++mt)
        #pragma unroll
        for (int nt = 0; nt < 4; ++nt) {
          const float bb = b1[l * 64 + nt * 16 + n15];
          #pragma unroll
          for (int r = 0; r < 8; ++r)
            hst[(mt * 16 + r + mhi) * 64 + nt * 16 + n15] =
                (h16)silu_f(a4[mt][nt][r] + bb);
        }
    }
    wave_lds_sync();
    // h = silu(h @ w2 + b2)
    {
      v8f a4[M_TILES][4] = {{{}, {}, {}, {}}, {{}, {}, {}, {}}};
      #pragma unroll
      for (int kt = 0; kt < 2; ++kt) {
        const v16h A0 = load_a_frag(hst, 64, kt * 32, lane);
        const v16h A1 = load_a_frag(hst + 16 * 64, 64, kt * 32, lane);
        #pragma unroll
        for (int nt = 0; nt < 4; ++nt) {
          const v16h B = load_b_frag(W2, kt * 4 + nt, lane);
          a4[0][nt] = wmma_f16(A0, B, a4[0][nt]);
          a4[1][nt] = wmma_f16(A1, B, a4[1][nt]);
        }
      }
      wave_lds_sync();  // reads of hst done before overwrite
      #pragma unroll
      for (int mt = 0; mt < M_TILES; ++mt)
        #pragma unroll
        for (int nt = 0; nt < 4; ++nt) {
          const float bb = b2[l * 64 + nt * 16 + n15];
          #pragma unroll
          for (int r = 0; r < 8; ++r)
            hst[(mt * 16 + r + mhi) * 64 + nt * 16 + n15] =
                (h16)silu_f(a4[mt][nt][r] + bb);
        }
    }
    wave_lds_sync();
    // ef = ef + (h @ w3 + b3)
    {
      v8f a8[M_TILES][8] = {{{}, {}, {}, {}, {}, {}, {}, {}},
                            {{}, {}, {}, {}, {}, {}, {}, {}}};
      #pragma unroll
      for (int kt = 0; kt < 2; ++kt) {
        const v16h A0 = load_a_frag(hst, 64, kt * 32, lane);
        const v16h A1 = load_a_frag(hst + 16 * 64, 64, kt * 32, lane);
        #pragma unroll
        for (int nt = 0; nt < 8; ++nt) {
          const v16h B = load_b_frag(W3, kt * 8 + nt, lane);
          a8[0][nt] = wmma_f16(A0, B, a8[0][nt]);
          a8[1][nt] = wmma_f16(A1, B, a8[1][nt]);
        }
      }
      #pragma unroll
      for (int mt = 0; mt < M_TILES; ++mt)
        #pragma unroll
        for (int nt = 0; nt < 8; ++nt) {
          const float bb = b3[l * 128 + nt * 16 + n15];
          #pragma unroll
          for (int r = 0; r < 8; ++r) {
            const int idx = (mt * 16 + r + mhi) * 288 + nt * 16 + n15;
            tile[idx] = (h16)(a8[mt][nt][r] + bb + (float)tile[idx]);
          }
        }
    }
    wave_lds_sync();
    // LayerNorm over the 128 ef features, one edge row per lane (32 rows)
    {
      h16* trow = tile + lane * 288;
      float mu = 0.0f;
      for (int c = 0; c < 128; ++c) mu += (float)trow[c];
      mu *= (1.0f / 128.0f);
      float var = 0.0f;
      for (int c = 0; c < 128; ++c) {
        const float t = (float)trow[c] - mu;
        var += t * t;
      }
      var *= (1.0f / 128.0f);
      const float rinv = rsqrtf(var + 1e-5f);
      for (int c = 0; c < 128; ++c) {
        const float v =
            ((float)trow[c] - mu) * rinv * ln_g[l * 128 + c] + ln_b[l * 128 + c];
        trow[c] = (h16)v;
      }
    }
    wave_lds_sync();
    // gate: w = silu(sh @ tpw1 + tpb1) @ tpw2 + tpb2 ; ef *= w
    {
      v8f t4[M_TILES][4] = {{{}, {}, {}, {}}, {{}, {}, {}, {}}};
      const v16h A0 = load_a_frag(tile, 288, 256, lane);  // rbf|sh|pad K-tile
      const v16h A1 = load_a_frag(tile + 16 * 288, 288, 256, lane);
      #pragma unroll
      for (int nt = 0; nt < 4; ++nt) {
        const v16h B = load_b_frag(T1, nt, lane);
        t4[0][nt] = wmma_f16(A0, B, t4[0][nt]);
        t4[1][nt] = wmma_f16(A1, B, t4[1][nt]);
      }
      #pragma unroll
      for (int mt = 0; mt < M_TILES; ++mt)
        #pragma unroll
        for (int nt = 0; nt < 4; ++nt) {
          const float bb = tpb1[l * 64 + nt * 16 + n15];
          #pragma unroll
          for (int r = 0; r < 8; ++r)
            hst[(mt * 16 + r + mhi) * 64 + nt * 16 + n15] =
                (h16)silu_f(t4[mt][nt][r] + bb);
        }
    }
    wave_lds_sync();
    {
      v8f g8[M_TILES][8] = {{{}, {}, {}, {}, {}, {}, {}, {}},
                            {{}, {}, {}, {}, {}, {}, {}, {}}};
      #pragma unroll
      for (int kt = 0; kt < 2; ++kt) {
        const v16h A0 = load_a_frag(hst, 64, kt * 32, lane);
        const v16h A1 = load_a_frag(hst + 16 * 64, 64, kt * 32, lane);
        #pragma unroll
        for (int nt = 0; nt < 8; ++nt) {
          const v16h B = load_b_frag(T2, kt * 8 + nt, lane);
          g8[0][nt] = wmma_f16(A0, B, g8[0][nt]);
          g8[1][nt] = wmma_f16(A1, B, g8[1][nt]);
        }
      }
      #pragma unroll
      for (int mt = 0; mt < M_TILES; ++mt)
        #pragma unroll
        for (int nt = 0; nt < 8; ++nt) {
          const float bb = tpb2[l * 128 + nt * 16 + n15];
          #pragma unroll
          for (int r = 0; r < 8; ++r) {
            const int idx = (mt * 16 + r + mhi) * 288 + nt * 16 + n15;
            tile[idx] = (h16)((float)tile[idx] * (g8[mt][nt][r] + bb));
          }
        }
    }
    wave_lds_sync();
  }

  // ---------------- head: 128 -> 64 -> 32 -> 1 ----------------
  {
    v8f p4[M_TILES][4] = {{{}, {}, {}, {}}, {{}, {}, {}, {}}};
    #pragma unroll
    for (int kt = 0; kt < 4; ++kt) {
      const v16h A0 = load_a_frag(tile, 288, kt * 32, lane);  // ef cols 0..127
      const v16h A1 = load_a_frag(tile + 16 * 288, 288, kt * 32, lane);
      #pragma unroll
      for (int nt = 0; nt < 4; ++nt) {
        const v16h B = load_b_frag(ws + OFF_HW1, kt * 4 + nt, lane);
        p4[0][nt] = wmma_f16(A0, B, p4[0][nt]);
        p4[1][nt] = wmma_f16(A1, B, p4[1][nt]);
      }
    }
    #pragma unroll
    for (int mt = 0; mt < M_TILES; ++mt)
      #pragma unroll
      for (int nt = 0; nt < 4; ++nt) {
        const float bb = hb1[nt * 16 + n15];
        #pragma unroll
        for (int r = 0; r < 8; ++r)
          hst[(mt * 16 + r + mhi) * 64 + nt * 16 + n15] =
              (h16)silu_f(p4[mt][nt][r] + bb);
      }
  }
  wave_lds_sync();
  {
    v8f q2[M_TILES][2] = {{{}, {}}, {{}, {}}};
    #pragma unroll
    for (int kt = 0; kt < 2; ++kt) {
      const v16h A0 = load_a_frag(hst, 64, kt * 32, lane);
      const v16h A1 = load_a_frag(hst + 16 * 64, 64, kt * 32, lane);
      #pragma unroll
      for (int nt = 0; nt < 2; ++nt) {
        const v16h B = load_b_frag(ws + OFF_HW2, kt * 2 + nt, lane);
        q2[0][nt] = wmma_f16(A0, B, q2[0][nt]);
        q2[1][nt] = wmma_f16(A1, B, q2[1][nt]);
      }
    }
    wave_lds_sync();
    #pragma unroll
    for (int mt = 0; mt < M_TILES; ++mt)
      #pragma unroll
      for (int nt = 0; nt < 2; ++nt) {
        const float bb = hb2[nt * 16 + n15];
        #pragma unroll
        for (int r = 0; r < 8; ++r)
          hst[(mt * 16 + r + mhi) * 64 + nt * 16 + n15] =
              (h16)silu_f(q2[mt][nt][r] + bb);
      }
  }
  wave_lds_sync();

  // final 32->1 dot, one edge row per lane
  float pe = 0.0f;
  if ((base_e + lane) < (long long)E) {
    const h16* hr = hst + lane * 64;
    float s = hb3[0];
    #pragma unroll
    for (int c = 0; c < 32; ++c) s += (float)hr[c] * hw3[c];
    pe = s;
  }
  #pragma unroll
  for (int off = 16; off > 0; off >>= 1) pe += __shfl_down(pe, off);
  if (lane == 0) atomicAdd(out, pe);
}

// --------------------------------------------------------------------------------
extern "C" void kernel_launch(void* const* d_in, const int* in_sizes, int n_in,
                              void* d_out, int out_size, void* d_ws, size_t ws_size,
                              hipStream_t stream) {
  (void)n_in; (void)out_size; (void)ws_size;
  const int*   atomic_numbers = (const int*)d_in[0];
  const float* pos      = (const float*)d_in[1];
  const int*   edge_idx = (const int*)d_in[2];
  const float* node_emb = (const float*)d_in[3];
  const float* atomic_e = (const float*)d_in[4];
  const float* w_init   = (const float*)d_in[5];
  const float* b_init   = (const float*)d_in[6];
  const float* w1  = (const float*)d_in[7];
  const float* b1  = (const float*)d_in[8];
  const float* w2  = (const float*)d_in[9];
  const float* b2  = (const float*)d_in[10];
  const float* w3  = (const float*)d_in[11];
  const float* b3  = (const float*)d_in[12];
  const float* lng = (const float*)d_in[13];
  const float* lnb = (const float*)d_in[14];
  const float* tpw1 = (const float*)d_in[15];
  const float* tpb1 = (const float*)d_in[16];
  const float* tpw2 = (const float*)d_in[17];
  const float* tpb2 = (const float*)d_in[18];
  const float* hw1 = (const float*)d_in[19];
  const float* hb1 = (const float*)d_in[20];
  const float* hw2 = (const float*)d_in[21];
  const float* hb2 = (const float*)d_in[22];
  const float* hw3 = (const float*)d_in[23];
  const float* hb3 = (const float*)d_in[24];

  const int N = in_sizes[0];
  const int E = in_sizes[2] / 2;
  h16*   ws  = (h16*)d_ws;
  float* out = (float*)d_out;

  zero_out_kernel<<<1, 1, 0, stream>>>(out);

  auto packLaunch = [&](const float* src, h16* dst, int Kr, int Nn, int Kp, int ro) {
    const int total  = Kp * Nn;
    const int blocks = (total + 255) / 256;
    pack_wmma_b<<<blocks, 256, 0, stream>>>(src, dst, Kr, Nn, Kp, ro);
  };
  packLaunch(w_init, ws + OFF_WINIT, 145, 128, 160, 0);
  for (int l = 0; l < 3; ++l) {
    packLaunch(w1 + l * 264 * 64,  ws + OFF_W1  + l * 18432, 264, 64, 288, 0);
    packLaunch(w2 + l * 64 * 64,   ws + OFF_W2  + l * 4096,  64,  64, 64,  0);
    packLaunch(w3 + l * 64 * 128,  ws + OFF_W3  + l * 8192,  64,  128, 64, 0);
    packLaunch(tpw1 + l * 9 * 64,  ws + OFF_TP1 + l * 2048,  9,   64, 32,  8);
    packLaunch(tpw2 + l * 64 * 128, ws + OFF_TP2 + l * 8192, 64,  128, 64, 0);
  }
  packLaunch(hw1, ws + OFF_HW1, 128, 64, 128, 0);
  packLaunch(hw2, ws + OFF_HW2, 64, 32, 64, 0);

  node_energy_kernel<<<256, 256, 0, stream>>>(atomic_numbers, atomic_e, out, N);

  const long long tiles  = ((long long)E + EDGES_PER_WAVE - 1) / EDGES_PER_WAVE;
  const int blocks = (int)((tiles + WAVES_PER_BLOCK - 1) / WAVES_PER_BLOCK);
  allegro_fused<<<blocks, WAVES_PER_BLOCK * 32, 0, stream>>>(
      atomic_numbers, pos, edge_idx, node_emb, ws, b_init, b1, b2, b3, lng, lnb,
      tpb1, tpb2, hb1, hb2, hw3, hb3, out, N, E);
}
